// SupConLoss_43035572306097
// MI455X (gfx1250) — compile-verified, matrix operand order checked by reading, not run on hardware
//
#include <hip/hip_runtime.h>
#include <hip/hip_bf16.h>

#define B_ROWS 8192
#define N_COLS 8192
#define D_K    256
#define NSPLIT 4
#define ROWS_PER_WG 128
#define COLS_PER_SPLIT (N_COLS / NSPLIT)   // 2048
#define COL_BLOCK 64
#define LDS_STRIDE 264   // ushorts per staged image row: 256 + 8 pad (bank-conflict free)

typedef __bf16 v16bf __attribute__((ext_vector_type(16)));
typedef __bf16 v8bf  __attribute__((ext_vector_type(8)));
typedef float  v8f   __attribute__((ext_vector_type(8)));

#if __has_builtin(__builtin_amdgcn_global_load_async_to_lds_b128)
#define HAVE_ASYNC_LDS 1
#else
#define HAVE_ASYNC_LDS 0
#endif

#if HAVE_ASYNC_LDS
typedef int v4i_b128 __attribute__((vector_size(16)));   // matches builtin param type
typedef __attribute__((address_space(1))) v4i_b128* gptr_b128;
typedef __attribute__((address_space(3))) v4i_b128* lptr_b128;
static __device__ __forceinline__ gptr_b128 as_gbl(const void* p) {
  return (gptr_b128)(unsigned long long)p;               // generic global addr == AS(1) addr
}
static __device__ __forceinline__ lptr_b128 as_lds(void* p) {
  // generic->LDS: LDS offset is the low 32 bits of the generic address
  return (lptr_b128)(unsigned int)(unsigned long long)p;
}
#endif

static __device__ __forceinline__ void wait_async0() {
#if HAVE_ASYNC_LDS
#if __has_builtin(__builtin_amdgcn_s_wait_asynccnt)
  __builtin_amdgcn_s_wait_asynccnt(0);
#else
  asm volatile("s_wait_asynccnt 0x0" ::: "memory");
#endif
#endif
}

// stage one 64-row x 256-K bf16 image block (32 KB) into LDS; 128 B per thread
static __device__ __forceinline__ void stage_block(unsigned short* lds_base,
                                                   const unsigned short* __restrict__ ibf,
                                                   int j0, int tid) {
  const int n = tid >> 2;   // 0..63 : staged row
  const int q = tid & 3;    // 128-byte quarter of the 512-byte row
  const unsigned short* src = ibf + (size_t)(j0 + n) * D_K + q * 64;
  unsigned short* dst = lds_base + n * LDS_STRIDE + q * 64;
#if HAVE_ASYNC_LDS
  __builtin_amdgcn_global_load_async_to_lds_b128(as_gbl(src), as_lds(dst), 0,   0);
  __builtin_amdgcn_global_load_async_to_lds_b128(as_gbl(src), as_lds(dst), 16,  0);
  __builtin_amdgcn_global_load_async_to_lds_b128(as_gbl(src), as_lds(dst), 32,  0);
  __builtin_amdgcn_global_load_async_to_lds_b128(as_gbl(src), as_lds(dst), 48,  0);
  __builtin_amdgcn_global_load_async_to_lds_b128(as_gbl(src), as_lds(dst), 64,  0);
  __builtin_amdgcn_global_load_async_to_lds_b128(as_gbl(src), as_lds(dst), 80,  0);
  __builtin_amdgcn_global_load_async_to_lds_b128(as_gbl(src), as_lds(dst), 96,  0);
  __builtin_amdgcn_global_load_async_to_lds_b128(as_gbl(src), as_lds(dst), 112, 0);
#else
  const uint4* s4 = (const uint4*)src;
  uint4* d4 = (uint4*)dst;
#pragma unroll
  for (int i = 0; i < 8; ++i) d4[i] = s4[i];
#endif
}

static __device__ __forceinline__ unsigned short f32_to_bf16_rne(float f) {
  unsigned int u = __float_as_uint(f);
  u += 0x7FFFu + ((u >> 16) & 1u);
  return (unsigned short)(u >> 16);
}

// merge two streaming base-2 logsumexp states (m, s): s = sum 2^(x - m)
static __device__ __forceinline__ void lse_combine(float& m, float& s, float om, float os) {
  float nm = fmaxf(m, om);
  float a = (m  == -__builtin_inff()) ? 0.f : s  * __builtin_amdgcn_exp2f(m  - nm);
  float b = (om == -__builtin_inff()) ? 0.f : os * __builtin_amdgcn_exp2f(om - nm);
  m = nm;
  s = a + b;
}

__global__ void convert_bf16_kernel(const float* __restrict__ text,
                                    const float* __restrict__ image,
                                    unsigned short* __restrict__ tbf,
                                    unsigned short* __restrict__ ibf) {
  int idx = blockIdx.x * blockDim.x + threadIdx.x;
  if (idx < B_ROWS * D_K) {
    tbf[idx] = f32_to_bf16_rne(text[idx]);
    ibf[idx] = f32_to_bf16_rne(image[idx]);   // N*D == B*D here
  }
}

__global__ __launch_bounds__(256)
void supcon_main_kernel(const unsigned short* __restrict__ tbf,
                        const unsigned short* __restrict__ ibf,
                        const int* __restrict__ tlab,
                        const int* __restrict__ itar,
                        float4* __restrict__ part) {
  __shared__ __align__(16) unsigned short lds_b[2][COL_BLOCK * LDS_STRIDE];

  const int tid  = threadIdx.x;
  const int wave = tid >> 5;
  const int lane = tid & 31;
  const int n16  = lane & 15;   // A: row-in-tile; B/C: col-in-tile
  const int h    = lane >> 4;   // half selector (K sub-block for A/B, M+8 for C)

  const int rb    = blockIdx.x;
  const int split = blockIdx.y;
  const int r0    = rb * ROWS_PER_WG + wave * 16;
  const int jbeg  = split * COLS_PER_SPLIT;

  // ---- A tile (16 rows x 256 K) held in registers for the whole kernel ----
  v16bf a[8];
  {
    const unsigned short* arow = tbf + (size_t)(r0 + n16) * D_K;
#pragma unroll
    for (int kc = 0; kc < 8; ++kc) {
      const v8bf* p = (const v8bf*)(arow + kc * 32 + h * 8);
      v8bf lo = p[0];           // K = kc*32 + h*8 + [0..7]
      v8bf hi = p[2];           // K = kc*32 + 16 + h*8 + [0..7]
      a[kc] = __builtin_shufflevector(lo, hi, 0,1,2,3,4,5,6,7,8,9,10,11,12,13,14,15);
    }
  }

  int lab[8];
#pragma unroll
  for (int v = 0; v < 8; ++v) lab[v] = tlab[r0 + v + 8 * h];   // C row m = v + 8h

  float mp[8], sp[8], mn[8], sn[8];
#pragma unroll
  for (int v = 0; v < 8; ++v) {
    mp[v] = -__builtin_inff(); sp[v] = 0.f;
    mn[v] = -__builtin_inff(); sn[v] = 0.f;
  }

  const float LOG2E = 1.4426950408889634f;
  const int nrounds = COLS_PER_SPLIT / COL_BLOCK;   // 32

  stage_block(&lds_b[0][0], ibf, jbeg, tid);        // prologue stage

  for (int r = 0; r < nrounds; ++r) {
    const int j0 = jbeg + r * COL_BLOCK;
    wait_async0();          // own async writes done (no-op on sync path)
    __syncthreads();        // everyone's writes visible; prev reads of 'next' done
    if (r + 1 < nrounds) stage_block(&lds_b[(r + 1) & 1][0], ibf, j0 + COL_BLOCK, tid);

    const unsigned short* cur = &lds_b[r & 1][0];
    int it4[4];
#pragma unroll
    for (int ct = 0; ct < 4; ++ct) it4[ct] = itar[j0 + ct * 16 + n16];

    v8f acc4[4];
#pragma unroll
    for (int ct = 0; ct < 4; ++ct) acc4[ct] = (v8f){0.f,0.f,0.f,0.f,0.f,0.f,0.f,0.f};

    // K loop: batch 8 ds_load_b128 (4 tiles x 2) into distinct regs, then 4 WMMAs
#pragma unroll
    for (int kc = 0; kc < 8; ++kc) {
      v16bf bfrag[4];
#pragma unroll
      for (int ct = 0; ct < 4; ++ct) {
        const v8bf* p = (const v8bf*)(cur + (ct * 16 + n16) * LDS_STRIDE + kc * 32 + h * 8);
        v8bf lo = p[0];
        v8bf hi = p[2];
        bfrag[ct] = __builtin_shufflevector(lo, hi, 0,1,2,3,4,5,6,7,8,9,10,11,12,13,14,15);
      }
#pragma unroll
      for (int ct = 0; ct < 4; ++ct) {
        acc4[ct] = __builtin_amdgcn_wmma_f32_16x16x32_bf16(
            false, a[kc], false, bfrag[ct], (short)0, acc4[ct], false, false);
      }
    }

    // branchless online base-2 LSE: 1 transcendental per element, no exec-mask
#pragma unroll
    for (int ct = 0; ct < 4; ++ct) {
      const int it = it4[ct];
#pragma unroll
      for (int v = 0; v < 8; ++v) {
        float x2 = acc4[ct][v] * LOG2E;        // TEMPERATURE == 1
        bool pos = (lab[v] == it);
        float xs = pos ? -x2 : x2;
        float m  = pos ? mp[v] : mn[v];
        float s  = pos ? sp[v] : sn[v];
        float d  = xs - m;                     // d > 0 -> new max
        float e  = __builtin_amdgcn_exp2f(-__builtin_fabsf(d));
        float s_new = (d > 0.f) ? __builtin_fmaf(s, e, 1.0f) : (s + e);
        float m_new = fmaxf(m, xs);            // m=-inf: e=0, s_new=1, m_new=xs  OK
        mp[v] = pos ? m_new : mp[v];
        sp[v] = pos ? s_new : sp[v];
        mn[v] = pos ? mn[v] : m_new;
        sn[v] = pos ? sn[v] : s_new;
      }
    }
  }

  // ---- reduce LSE states across the 16 lanes sharing each output row ----
#pragma unroll
  for (int v = 0; v < 8; ++v) {
#pragma unroll
    for (int off = 1; off < 16; off <<= 1) {
      float om = __shfl_xor(mp[v], off, 32);
      float os = __shfl_xor(sp[v], off, 32);
      lse_combine(mp[v], sp[v], om, os);
      om = __shfl_xor(mn[v], off, 32);
      os = __shfl_xor(sn[v], off, 32);
      lse_combine(mn[v], sn[v], om, os);
    }
  }

  if (n16 == 0) {
#pragma unroll
    for (int v = 0; v < 8; ++v) {
      part[(size_t)split * B_ROWS + r0 + v + 8 * h] =
          make_float4(mp[v], sp[v], mn[v], sn[v]);
    }
  }
}

__global__ void finalize_kernel(const float4* __restrict__ part,
                                float* __restrict__ blocksum) {
  __shared__ float red[256];
  const int row = blockIdx.x * 256 + threadIdx.x;
  float mpv = -__builtin_inff(), spv = 0.f, mnv = -__builtin_inff(), snv = 0.f;
#pragma unroll
  for (int k = 0; k < NSPLIT; ++k) {
    float4 p = part[(size_t)k * B_ROWS + row];
    lse_combine(mpv, spv, p.x, p.y);
    lse_combine(mnv, snv, p.z, p.w);
  }
  const float LN2 = 0.6931471805599453f;
  float lsep = (mpv + log2f(spv)) * LN2;
  float lsen = (mnv + log2f(snv)) * LN2;
  float z = lsen + lsep;
  float l = (z > 20.f) ? z : log1pf(__expf(z));   // stable softplus; z=-inf -> 0
  red[threadIdx.x] = l;
  __syncthreads();
  for (int s = 128; s > 0; s >>= 1) {
    if (threadIdx.x < s) red[threadIdx.x] += red[threadIdx.x + s];
    __syncthreads();
  }
  if (threadIdx.x == 0) blocksum[blockIdx.x] = red[0];
}

__global__ void final_sum_kernel(const float* __restrict__ blocksum,
                                 float* __restrict__ out) {
  float v = blocksum[threadIdx.x];            // 32 partials, single wave
#pragma unroll
  for (int off = 16; off > 0; off >>= 1) v += __shfl_down(v, off, 32);
  if (threadIdx.x == 0) out[0] = v * (1.0f / (float)B_ROWS);
}

extern "C" void kernel_launch(void* const* d_in, const int* in_sizes, int n_in,
                              void* d_out, int out_size, void* d_ws, size_t ws_size,
                              hipStream_t stream) {
  const float* text  = (const float*)d_in[0];
  const float* image = (const float*)d_in[1];
  const int*   tlab  = (const int*)d_in[2];
  const int*   itar  = (const int*)d_in[3];
  float* out = (float*)d_out;

  unsigned short* tbf = (unsigned short*)d_ws;                       // 4 MB
  unsigned short* ibf = tbf + (size_t)B_ROWS * D_K;                  // 4 MB
  float4* part = (float4*)(ibf + (size_t)N_COLS * D_K);              // 512 KB
  float* blocksum = (float*)(part + (size_t)NSPLIT * B_ROWS);        // 128 B

  convert_bf16_kernel<<<(B_ROWS * D_K + 255) / 256, 256, 0, stream>>>(text, image, tbf, ibf);
  supcon_main_kernel<<<dim3(B_ROWS / ROWS_PER_WG, NSPLIT), 256, 0, stream>>>(
      tbf, ibf, tlab, itar, part);
  finalize_kernel<<<B_ROWS / 256, 256, 0, stream>>>(part, blocksum);
  final_sum_kernel<<<1, 32, 0, stream>>>(blocksum, out);
}
